// _ParityGameGATConv_27075473834771
// MI455X (gfx1250) — matrix operand order repaired and checked by reading, not compiled, not run on hardware
//
#include <hip/hip_runtime.h>

// ParityGame GAT pipeline for gfx1250 (MI455X, wave32, WMMA).
// v2: all WMMA fragments are loaded with wide (b128) LDS/global accesses:
//  - weights stored TRANSPOSED in LDS so B-fragments are K-contiguous,
//  - A-fragments are two contiguous 16B runs,
//  - conflict-free padded strides (ldk = K+8, stage stride 72/168).

#define NNODES 100000
#define NEDGES 1600000
#define DIN    128
#define HDIM   64
#define DE     8
#define NC     2
#define KPAD   160     // 136 edge-feature K padded up for 32-wide WMMA chunks
#define KSTR_E 168     // edge stage / We1T K-stride (multiple of 8, bank-rotating)
#define LDK_ENC 136    // Wenc  transposed K-stride (128+8)
#define LDK_H   72     // 64+8 for Wgat / Wn1 transposed and h-stage

typedef __attribute__((ext_vector_type(16))) _Float16 v16h;
typedef __attribute__((ext_vector_type(8)))  _Float16 v8h;
typedef __attribute__((ext_vector_type(2)))  _Float16 v2h;
typedef __attribute__((ext_vector_type(8)))  float    v8f;
typedef __attribute__((ext_vector_type(4)))  float    v4f;
typedef __attribute__((ext_vector_type(2)))  float    v2f;

__device__ __forceinline__ v8f wmma_f16(v16h a, v16h b, v8f c) {
  return __builtin_amdgcn_wmma_f32_16x16x32_f16(false, a, false, b, (short)0, c,
                                                false, false);
}

__device__ __forceinline__ v16h cat8(v8h lo, v8h hi) {
  return __builtin_shufflevector(lo, hi, 0, 1, 2, 3, 4, 5, 6, 7, 8, 9, 10, 11,
                                 12, 13, 14, 15);
}

// ---- fragment builders (ISA 7.12.2 layouts, wave32) -----------------------
// A (16x32 f16) element e -> K = k0 + 16*(e>>3) + 8*hh + (e&7):
// two contiguous 8-element (16B) runs at K = k0+8hh and k0+16+8hh.
__device__ __forceinline__ v16h a_frag_lds(const _Float16* s, int stride,
                                           int lane, int k0) {
  const int m = lane & 15, hh = lane >> 4;
  const _Float16* p = s + m * stride + k0 + 8 * hh;
  return cat8(*(const v8h*)p, *(const v8h*)(p + 16));
}

// B (32x16 f16) element e -> K = k0 + 16*(e>>3) + 8*hh + (e&7) for column n:
// weights stored transposed [64][ldk] give two 16B runs (same order as A).
__device__ __forceinline__ v16h b_frag_ldsT(const _Float16* wT, int ldk,
                                            int lane, int k0, int n0) {
  const _Float16* p = wT + (n0 + (lane & 15)) * ldk + k0 + 8 * (lane >> 4);
  return cat8(*(const v8h*)p, *(const v8h*)(p + 16));
}

// A fragment straight from global f32 rows: 4x float4 (global_load_b128).
__device__ __forceinline__ v16h a_frag_gmem_f32(const float* X, long row0,
                                                int ld, int lane, int k0,
                                                bool relu) {
  const int m = lane & 15, hh = lane >> 4;
  const float* row = X + (row0 + (long)m) * (long)ld + k0 + 8 * hh;
  v4f f0 = *(const v4f*)(row);
  v4f f1 = *(const v4f*)(row + 4);
  v4f f2 = *(const v4f*)(row + 16);
  v4f f3 = *(const v4f*)(row + 20);
  v16h a;
#pragma unroll
  for (int e = 0; e < 4; ++e) {
    float g0 = f0[e], g1 = f1[e], g2 = f2[e], g3 = f3[e];
    if (relu) {
      g0 = fmaxf(g0, 0.f); g1 = fmaxf(g1, 0.f);
      g2 = fmaxf(g2, 0.f); g3 = fmaxf(g3, 0.f);
    }
    a[e] = (_Float16)g0; a[e + 4] = (_Float16)g1;
    a[e + 8] = (_Float16)g2; a[e + 12] = (_Float16)g3;
  }
  return a;
}

// monotonic float<->int key (for atomicMax segment-max on signed int)
__device__ __forceinline__ int fkey(float f) {
  int i = __float_as_int(f);
  return i >= 0 ? i : (i ^ 0x7FFFFFFF);
}
__device__ __forceinline__ float unfkey(int k) {
  return __int_as_float(k >= 0 ? k : (k ^ 0x7FFFFFFF));
}

// ---- kernel 0: workspace init ---------------------------------------------
__global__ __launch_bounds__(256) void k_init(float* __restrict__ agg,
                                              float* __restrict__ denom,
                                              int* __restrict__ mkey) {
  const long i = (long)blockIdx.x * 256 + threadIdx.x;   // exact: N*HDIM/256
  agg[i] = 0.f;
  if (i < NNODES) { denom[i] = 0.f; mkey[i] = 0x807FFFFF; /* key(-inf) */ }
}

// ---- kernel 1: hg = relu(x@Wenc + b) @ Wgat  (fused, WMMA) ----------------
__global__ __launch_bounds__(64) void k_encode_core(
    const float* __restrict__ x, const float* __restrict__ Wenc,
    const float* __restrict__ benc, const float* __restrict__ Wgat,
    float* __restrict__ hg) {
  __shared__ __align__(16) _Float16 sWencT[HDIM * LDK_ENC];  // transposed
  __shared__ __align__(16) _Float16 sWgatT[HDIM * LDK_H];    // transposed
  __shared__ float sB[HDIM];
  __shared__ __align__(16) _Float16 sStage[2][16 * LDK_H];   // per-wave h tile

  const int tid = threadIdx.x;
  for (int i = tid; i < DIN * HDIM; i += 64) {               // coalesced read
    const int k = i / HDIM, c = i % HDIM;
    sWencT[c * LDK_ENC + k] = (_Float16)Wenc[i];
  }
  for (int i = tid; i < HDIM * HDIM; i += 64) {
    const int k = i / HDIM, c = i % HDIM;
    sWgatT[c * LDK_H + k] = (_Float16)Wgat[i];
  }
  if (tid < HDIM) sB[tid] = benc[tid];
  __syncthreads();

  const int wave = tid >> 5, lane = tid & 31;
  const long tile = (long)blockIdx.x * 2 + wave;   // 3125 blocks -> 6250 tiles
  const long row0 = tile * 16;
  const int n = lane & 15, hh = lane >> 4;

  v8f acc[4] = {};
#pragma unroll
  for (int k0 = 0; k0 < DIN; k0 += 32) {
    v16h a = a_frag_gmem_f32(x, row0, DIN, lane, k0, false);
#pragma unroll
    for (int t = 0; t < 4; ++t)
      acc[t] = wmma_f16(a, b_frag_ldsT(sWencT, LDK_ENC, lane, k0, t * 16),
                        acc[t]);
  }
  _Float16* stg = sStage[wave];
#pragma unroll
  for (int t = 0; t < 4; ++t) {
    const float bias = sB[t * 16 + n];
#pragma unroll
    for (int v = 0; v < 8; ++v) {
      float val = acc[t][v] + bias;
      val = val > 0.f ? val : 0.f;
      stg[(v + 8 * hh) * LDK_H + t * 16 + n] = (_Float16)val;
    }
  }
  __syncthreads();

  v8f acc2[4] = {};
#pragma unroll
  for (int k0 = 0; k0 < HDIM; k0 += 32) {
    v16h a = a_frag_lds(stg, LDK_H, lane, k0);
#pragma unroll
    for (int t = 0; t < 4; ++t)
      acc2[t] = wmma_f16(a, b_frag_ldsT(sWgatT, LDK_H, lane, k0, t * 16),
                         acc2[t]);
  }
#pragma unroll
  for (int t = 0; t < 4; ++t)
#pragma unroll
    for (int v = 0; v < 8; ++v)
      hg[(row0 + v + 8 * hh) * HDIM + t * 16 + n] = acc2[t][v];
}

// ---- kernel 2: per-node attention scalars ---------------------------------
__global__ __launch_bounds__(256) void k_node_scores(
    const float* __restrict__ hg, const float* __restrict__ a_src,
    const float* __restrict__ a_dst, float* __restrict__ ssrc,
    float* __restrict__ sdst) {
  const int wave = threadIdx.x >> 5, lane = threadIdx.x & 31;
  const long node = (long)blockIdx.x * 8 + wave;   // 12500 blocks -> N waves
  const float* row = hg + node * HDIM;
  const float r0 = row[lane], r1 = row[lane + 32];
  float p0 = r0 * a_src[lane] + r1 * a_src[lane + 32];
  float p1 = r0 * a_dst[lane] + r1 * a_dst[lane + 32];
#pragma unroll
  for (int o = 16; o > 0; o >>= 1) {
    p0 += __shfl_xor(p0, o);
    p1 += __shfl_xor(p1, o);
  }
  if (lane == 0) { ssrc[node] = p0; sdst[node] = p1; }
}

// ---- kernel 3: edge scores + segment max (atomic int-key) -----------------
__global__ __launch_bounds__(256) void k_edge_score_max(
    const int* __restrict__ ei, const float* __restrict__ ssrc,
    const float* __restrict__ sdst, float* __restrict__ score,
    int* __restrict__ mkey) {
  const long e = (long)blockIdx.x * 256 + threadIdx.x;   // exact grid
  const int s = ei[e], d = ei[NEDGES + e];
  float sc = ssrc[s] + sdst[d];
  sc = sc > 0.f ? sc : 0.2f * sc;                        // leaky_relu(0.2)
  score[e] = sc;
  atomicMax(&mkey[d], fkey(sc));
}

// ---- kernel 4: decode max, guard isolated nodes ---------------------------
__global__ __launch_bounds__(256) void k_decode_max(const int* __restrict__ mkey,
                                                    float* __restrict__ m) {
  const long nid = (long)blockIdx.x * 256 + threadIdx.x;
  if (nid >= NNODES) return;
  float v = unfkey(mkey[nid]);
  if (isinf(v) && v < 0.f) v = 0.f;
  m[nid] = v;
}

// ---- kernel 5: ex = exp(score - m[dst]); denom[dst] += ex -----------------
__global__ __launch_bounds__(256) void k_edge_exp(
    const int* __restrict__ ei, const float* __restrict__ m,
    float* __restrict__ score, float* __restrict__ denom) {
  const long e = (long)blockIdx.x * 256 + threadIdx.x;   // exact grid
  const int d = ei[NEDGES + e];
  const float ex = __expf(score[e] - m[d]);
  score[e] = ex;                                         // reuse buffer
  atomicAdd(&denom[d], ex);
}

// ---- kernel 6: agg[dst] += alpha * hg[src]  (wave per edge) ---------------
__global__ __launch_bounds__(256) void k_edge_agg(
    const int* __restrict__ ei, const float* __restrict__ ex,
    const float* __restrict__ denom, const float* __restrict__ hg,
    float* __restrict__ agg) {
  const int wave = threadIdx.x >> 5, lane = threadIdx.x & 31;
  const long e = (long)blockIdx.x * 8 + wave;            // 200000 blocks
  const int s = ei[e], d = ei[NEDGES + e];
  const float alpha = ex[e] / (denom[d] + 1e-9f);
  const float* hs = hg + (long)s * HDIM;
  float* ag = agg + (long)d * HDIM;
  atomicAdd(&ag[lane], alpha * hs[lane]);
  atomicAdd(&ag[lane + 32], alpha * hs[lane + 32]);
}

// ---- kernel 7: xc = relu(agg) (in place) + node head WMMA -----------------
__global__ __launch_bounds__(64) void k_node_head(
    float* __restrict__ aggxc, const float* __restrict__ Wn1,
    const float* __restrict__ bn1, const float* __restrict__ Wn2,
    const float* __restrict__ bn2, float* __restrict__ node_logits) {
  __shared__ __align__(16) _Float16 sW1T[HDIM * LDK_H];   // transposed
  __shared__ float sW2[HDIM * NC];
  __shared__ float sB1[HDIM];
  __shared__ float sB2[NC];
  __shared__ __align__(16) float sMid[2][16 * 68];

  const int tid = threadIdx.x;
  for (int i = tid; i < HDIM * HDIM; i += 64) {
    const int k = i / HDIM, c = i % HDIM;
    sW1T[c * LDK_H + k] = (_Float16)Wn1[i];
  }
  for (int i = tid; i < HDIM * NC; i += 64) sW2[i] = Wn2[i];
  if (tid < HDIM) sB1[tid] = bn1[tid];
  if (tid < NC) sB2[tid] = bn2[tid];
  __syncthreads();

  const int wave = tid >> 5, lane = tid & 31;
  const long tile = (long)blockIdx.x * 2 + wave;   // 3125 blocks -> 6250 tiles
  const long row0 = tile * 16;
  const int n = lane & 15, hh = lane >> 4;

  // materialize xc = relu(agg) in place (rows owned exclusively by this wave;
  // A-fragments below re-apply relu on load, so ordering is benign)
  for (int i = lane; i < 16 * HDIM / 4; i += 32) {
    v4f* p = (v4f*)(aggxc + row0 * HDIM) + i;
    v4f v = *p;
#pragma unroll
    for (int j = 0; j < 4; ++j) v[j] = fmaxf(v[j], 0.f);
    *p = v;
  }

  v8f acc[4] = {};
#pragma unroll
  for (int k0 = 0; k0 < HDIM; k0 += 32) {
    v16h a = a_frag_gmem_f32(aggxc, row0, HDIM, lane, k0, true);
#pragma unroll
    for (int t = 0; t < 4; ++t)
      acc[t] = wmma_f16(a, b_frag_ldsT(sW1T, LDK_H, lane, k0, t * 16), acc[t]);
  }
  float* mid = sMid[wave];
#pragma unroll
  for (int t = 0; t < 4; ++t) {
    const float bias = sB1[t * 16 + n];
#pragma unroll
    for (int v = 0; v < 8; ++v) {
      const float val = acc[t][v] + bias;
      mid[(v + 8 * hh) * 68 + t * 16 + n] = val > 0.f ? val : 0.f;
    }
  }
  __syncthreads();
  {
    const int row = lane & 15, c = lane >> 4;
    float sum = sB2[c];
#pragma unroll
    for (int k = 0; k < HDIM; k += 2) {
      v2f mv = *(const v2f*)&mid[row * 68 + k];
      sum += mv.x * sW2[k * NC + c] + mv.y * sW2[(k + 1) * NC + c];
    }
    node_logits[(row0 + row) * NC + c] = sum;
  }
}

// ---- kernel 8: edge head — the dominant [1.6M,136]@[136,64] GEMM ----------
__global__ __launch_bounds__(64) void k_edge_head(
    const int* __restrict__ ei, const float* __restrict__ xc,
    const float* __restrict__ ea, const float* __restrict__ We1,
    const float* __restrict__ be1, const float* __restrict__ We2,
    const float* __restrict__ be2, float* __restrict__ edge_logits) {
  __shared__ __align__(16) _Float16 sW1T[HDIM * KSTR_E];  // transposed, 0-pad
  __shared__ float sW2[HDIM * NC];
  __shared__ float sB1[HDIM];
  __shared__ float sB2[NC];
  __shared__ __align__(16) unsigned char sScratch[2][16 * KSTR_E * 2];

  const int tid = threadIdx.x;
  for (int i = tid; i < KPAD * HDIM; i += 64) {
    const int k = i / HDIM, c = i % HDIM;
    sW1T[c * KSTR_E + k] =
        (k < (2 * HDIM + DE)) ? (_Float16)We1[k * HDIM + c] : (_Float16)0.f;
  }
  for (int i = tid; i < HDIM * NC; i += 64) sW2[i] = We2[i];
  if (tid < HDIM) sB1[tid] = be1[tid];
  if (tid < NC) sB2[tid] = be2[tid];
  __syncthreads();

  const int wave = tid >> 5, lane = tid & 31;
  const long tile = (long)blockIdx.x * 2 + wave;  // 50000 blocks -> 100000 tiles
  const long e0 = tile * 16;
  const int n = lane & 15, hh = lane >> 4;

  // stage 16 edges' features [xc[src] | xc[dst] | edge_attr | 0], K pairs:
  // all boundaries (64/128/136/160) are even -> b64 gathers + b32 LDS stores.
  _Float16* stg = (_Float16*)sScratch[wave];      // [16][KSTR_E]
  for (int i = lane; i < 16 * (KPAD / 2); i += 32) {
    const int r = i / (KPAD / 2);
    const int kk = (i % (KPAD / 2)) * 2;
    const long e = e0 + r;
    v2f v;
    if (kk < HDIM)              v = *(const v2f*)&xc[(long)ei[e] * HDIM + kk];
    else if (kk < 2 * HDIM)     v = *(const v2f*)&xc[(long)ei[NEDGES + e] * HDIM + (kk - HDIM)];
    else if (kk < 2 * HDIM + DE) v = *(const v2f*)&ea[e * DE + (kk - 2 * HDIM)];
    else                         v = (v2f){0.f, 0.f};
    *(v2h*)&stg[r * KSTR_E + kk] = (v2h){(_Float16)v.x, (_Float16)v.y};
  }
  __syncthreads();

  v8f acc[4] = {};
#pragma unroll
  for (int k0 = 0; k0 < KPAD; k0 += 32) {
    v16h a = a_frag_lds(stg, KSTR_E, lane, k0);
#pragma unroll
    for (int t = 0; t < 4; ++t)
      acc[t] = wmma_f16(a, b_frag_ldsT(sW1T, KSTR_E, lane, k0, t * 16), acc[t]);
  }
  __syncthreads();   // all A-frag LDS reads done before scratch is reused

  float* mid = (float*)sScratch[wave];            // [16][68] fp32 (4352B)
#pragma unroll
  for (int t = 0; t < 4; ++t) {
    const float bias = sB1[t * 16 + n];
#pragma unroll
    for (int v = 0; v < 8; ++v) {
      const float val = acc[t][v] + bias;
      mid[(v + 8 * hh) * 68 + t * 16 + n] = val > 0.f ? val : 0.f;
    }
  }
  __syncthreads();
  {
    const int row = lane & 15, c = lane >> 4;
    float sum = sB2[c];
#pragma unroll
    for (int k = 0; k < HDIM; k += 2) {
      v2f mv = *(const v2f*)&mid[row * 68 + k];
      sum += mv.x * sW2[k * NC + c] + mv.y * sW2[(k + 1) * NC + c];
    }
    edge_logits[(e0 + row) * NC + c] = sum;
  }
}

// ---- host launcher --------------------------------------------------------
extern "C" void kernel_launch(void* const* d_in, const int* in_sizes, int n_in,
                              void* d_out, int out_size, void* d_ws,
                              size_t ws_size, hipStream_t stream) {
  (void)in_sizes; (void)n_in; (void)out_size; (void)ws_size;

  const float* x    = (const float*)d_in[0];
  const int*   ei   = (const int*)d_in[1];
  const float* ea   = (const float*)d_in[2];
  const float* Wenc = (const float*)d_in[3];
  const float* benc = (const float*)d_in[4];
  const float* Wgat = (const float*)d_in[5];
  const float* asrc = (const float*)d_in[6];
  const float* adst = (const float*)d_in[7];
  const float* Wn1  = (const float*)d_in[8];
  const float* bn1  = (const float*)d_in[9];
  const float* Wn2  = (const float*)d_in[10];
  const float* bn2  = (const float*)d_in[11];
  const float* We1  = (const float*)d_in[12];
  const float* be1  = (const float*)d_in[13];
  const float* We2  = (const float*)d_in[14];
  const float* be2  = (const float*)d_in[15];

  float* node_logits = (float*)d_out;                          // [N, 2]
  float* edge_logits = node_logits + (long)NNODES * NC;        // [E, 2]

  char* ws = (char*)d_ws;
  size_t off = 0;
  auto take = [&](size_t bytes) {
    void* p = ws + off;
    off += (bytes + 255) & ~(size_t)255;
    return p;
  };
  float* hg    = (float*)take(sizeof(float) * (size_t)NNODES * HDIM);
  float* agg   = (float*)take(sizeof(float) * (size_t)NNODES * HDIM);  // ->xc
  float* score = (float*)take(sizeof(float) * (size_t)NEDGES);         // ->ex
  float* ssrc  = (float*)take(sizeof(float) * (size_t)NNODES);
  float* sdst  = (float*)take(sizeof(float) * (size_t)NNODES);
  float* mval  = (float*)take(sizeof(float) * (size_t)NNODES);
  float* denom = (float*)take(sizeof(float) * (size_t)NNODES);
  int*   mkey  = (int*)take(sizeof(int) * (size_t)NNODES);

  k_init<<<(NNODES * HDIM) / 256, 256, 0, stream>>>(agg, denom, mkey);
  k_encode_core<<<3125, 64, 0, stream>>>(x, Wenc, benc, Wgat, hg);
  k_node_scores<<<12500, 256, 0, stream>>>(hg, asrc, adst, ssrc, sdst);
  k_edge_score_max<<<NEDGES / 256, 256, 0, stream>>>(ei, ssrc, sdst, score, mkey);
  k_decode_max<<<(NNODES + 255) / 256, 256, 0, stream>>>(mkey, mval);
  k_edge_exp<<<NEDGES / 256, 256, 0, stream>>>(ei, mval, score, denom);
  k_edge_agg<<<NEDGES / 8, 256, 0, stream>>>(ei, score, denom, hg, agg);
  k_node_head<<<3125, 64, 0, stream>>>(agg, Wn1, bn1, Wn2, bn2, node_logits);
  k_edge_head<<<50000, 64, 0, stream>>>(ei, agg, ea, We1, be1, We2, be2,
                                        edge_logits);
}
